// cft_32899449487592
// MI455X (gfx1250) — matrix operand order repaired and checked by reading, not compiled, not run on hardware
//
#include <hip/hip_runtime.h>

// ---------------------------------------------------------------------------
// Types for CDNA5 WMMA (wave32, v_wmma_f32_16x16x32_bf16)
// ---------------------------------------------------------------------------
typedef __attribute__((ext_vector_type(16))) __bf16 v16bf;
typedef __attribute__((ext_vector_type(8)))  __bf16 v8bf;
typedef __attribute__((ext_vector_type(8)))  float  v8f;

union FragU { v16bf v; v8bf h[2]; };

// A-matrix fragment (16x32 bf16, M x K), LDS tile stored row-major [16][32].
// Lane L: m = L%16, half = L/16. Elements 0..7 -> K = half*8 + 0..7,
// elements 8..15 -> K = 16 + half*8 + 0..7  (ISA 7.12.2 16-bit A layout).
__device__ inline v16bf frag_a(const __bf16* tile, int lane) {
  int m = lane & 15, hf = (lane >> 4) & 1;
  const __bf16* row = tile + m * 32;
  FragU u;
  u.h[0] = *(const v8bf*)(row + hf * 8);
  u.h[1] = *(const v8bf*)(row + 16 + hf * 8);
  return u.v;
}

// B-matrix fragment (32x16 bf16, K x N), LDS tile stored N-major [16][32]
// (tile[n][k]).  Lane L: n = L%16, half = L/16. Elements 0..7 -> K = half*8 + 0..7,
// elements 8..15 -> K = 16 + half*8 + 0..7 (same as A).
__device__ inline v16bf frag_b(const __bf16* tile, int lane) {
  int n = lane & 15, hf = (lane >> 4) & 1;
  const __bf16* row = tile + n * 32;
  FragU u;
  u.h[0] = *(const v8bf*)(row + hf * 8);
  u.h[1] = *(const v8bf*)(row + 16 + hf * 8);
  return u.v;
}

__device__ inline v8f wmma_bf16(v16bf a, v16bf b, v8f c) {
  return __builtin_amdgcn_wmma_f32_16x16x32_bf16(
      /*neg_a=*/false, a, /*neg_b=*/false, b,
      /*c_mod=*/(short)0, c, /*reuse_a=*/false, /*reuse_b=*/false);
}

__device__ inline float gelu_exact(float x) {
  return 0.5f * x * (1.0f + erff(x * 0.70710678118654752f));
}

// Async global->LDS 16B copy (CDNA5 GLOBAL_LOAD_ASYNC_TO_LDS_B128, ASYNCcnt).
// ldsDst: wave-relative LDS byte address (low 32 bits of the generic pointer,
// per ISA flat->LDS aperture truncation).  Fallback: VGPR round-trip copy.
__device__ inline void async_copy_b128(void* ldsDst, const void* gsrc) {
#if defined(__gfx1250__)
  unsigned lds = (unsigned)(unsigned long long)ldsDst;
  asm volatile("global_load_async_to_lds_b128 %0, %1, off"
               :: "v"(lds), "v"(gsrc)
               : "memory");
#else
  *(v8bf*)ldsDst = *(const v8bf*)gsrc;
#endif
}

__device__ inline void async_wait() {
#if defined(__gfx1250__)
  asm volatile("s_wait_asynccnt 0x0" ::: "memory");
#endif
}

// Problem constants
#define BB   8
#define CC   256
#define HH   64
#define WW   64
#define PIX  4096          // H*W
#define NTOT 32768         // B*H*W
#define HID  1024
#define C8   32

// ---------------------------------------------------------------------------
// Generic pixel-column GEMM:  Out[Cout x N] = W[Cout x Cin] * X[Cin x N]
// X / Out layout: [b][C][4096] with pixel fastest (NCHW).
// MODE 0: out_bf16 = acc + bias
// MODE 1: out_bf16 = gelu(acc + bias)
// MODE 2: out_f32  = acc + bias + resid   (final residual store)
// ---------------------------------------------------------------------------
template <int MT, int NT, int WM, int WN, int MODE>
__global__ __launch_bounds__(WM * WN * 32)
void gemm_pix(const __bf16* __restrict__ Wgt, int Cout, int Cin,
              const __bf16* __restrict__ X,
              const float* __restrict__ bias,
              __bf16* __restrict__ Obf,
              float* __restrict__ Of32,
              const float* __restrict__ resid) {
  constexpr int THREADS = WM * WN * 32;
  constexpr int MF = MT / (16 * WM);
  constexpr int NF = NT / (16 * WN);

  __shared__ __bf16 sA[MT * 32];
  __shared__ __bf16 sB[NT * 32];

  const int tid  = threadIdx.x;
  const int lane = tid & 31;
  const int wave = tid >> 5;
  const int wm   = wave % WM;
  const int wn   = wave / WM;

  const int n0   = blockIdx.x * NT;
  const int m0   = blockIdx.y * MT;
  const int bimg = n0 >> 12;
  const int pix  = n0 & 4095;
  const __bf16* Xcol = X + (size_t)bimg * Cin * PIX + pix;

  v8f acc[MF][NF];
#pragma unroll
  for (int i = 0; i < MF; ++i)
#pragma unroll
    for (int j = 0; j < NF; ++j) {
      v8f z = {0.f, 0.f, 0.f, 0.f, 0.f, 0.f, 0.f, 0.f};
      acc[i][j] = z;
    }

  for (int k0 = 0; k0 < Cin; k0 += 32) {
    // Stage A tile [MT][32] via async DMA (no VGPR round trip).
    for (int idx = tid * 8; idx < MT * 32; idx += THREADS * 8) {
      int r = idx >> 5, c = idx & 31;
      async_copy_b128(sA + idx, Wgt + (size_t)(m0 + r) * Cin + k0 + c);
    }
    // Stage B tile transposed into [n][k]; global rows are contiguous in n.
    for (int idx = tid * 8; idx < NT * 32; idx += THREADS * 8) {
      int k  = idx / NT;
      int nn = idx % NT;
      v8bf ch = *(const v8bf*)(Xcol + (size_t)(k0 + k) * PIX + nn);
#pragma unroll
      for (int j = 0; j < 8; ++j) sB[(nn + j) * 32 + k] = ch[j];
    }
    async_wait();          // ASYNCcnt == 0: A tile resident in LDS
    __syncthreads();

    if (k0 + 32 < Cin) {  // prefetch next B k-slab (global_prefetch_b8)
      __builtin_prefetch(
          Xcol + (size_t)(k0 + 32 + (tid >> 3)) * PIX + ((tid & 7) << 4), 0, 0);
    }

    v16bf aF[MF], bF[NF];
#pragma unroll
    for (int i = 0; i < MF; ++i)
      aF[i] = frag_a(sA + (wm * (MT / WM) + i * 16) * 32, lane);
#pragma unroll
    for (int j = 0; j < NF; ++j)
      bF[j] = frag_b(sB + (wn * (NT / WN) + j * 16) * 32, lane);
#pragma unroll
    for (int i = 0; i < MF; ++i)
#pragma unroll
      for (int j = 0; j < NF; ++j)
        acc[i][j] = wmma_bf16(aF[i], bF[j], acc[i][j]);
    __syncthreads();
  }

  // Epilogue.  C/D layout: VGPR v -> row m = v + (lane/16)*8, col n = lane%16.
  const int hf = lane >> 4, nl = lane & 15;
  const size_t obase = (size_t)bimg * Cout * PIX + pix;
#pragma unroll
  for (int i = 0; i < MF; ++i)
#pragma unroll
    for (int j = 0; j < NF; ++j) {
      int ncol = wn * (NT / WN) + j * 16 + nl;
#pragma unroll
      for (int v = 0; v < 8; ++v) {
        int m = m0 + wm * (MT / WM) + i * 16 + hf * 8 + v;
        float val = acc[i][j][v] + (bias ? bias[m] : 0.f);
        size_t addr = obase + (size_t)m * PIX + ncol;
        if constexpr (MODE == 0) {
          Obf[addr] = (__bf16)val;
        } else if constexpr (MODE == 1) {
          Obf[addr] = (__bf16)gelu_exact(val);
        } else {
          Of32[addr] = val + resid[addr];
        }
      }
    }
}

// ---------------------------------------------------------------------------
// Small prep kernels
// ---------------------------------------------------------------------------
__global__ void k_prep_bn(const float* g1, const float* b1, const float* m1,
                          const float* v1, const float* g2, const float* b2,
                          const float* m2, const float* v2, float* s1,
                          float* t1, float* s2, float* t2) {
  int c = threadIdx.x;
  float a = g1[c] * rsqrtf(v1[c] + 1e-5f);
  s1[c] = a;
  t1[c] = b1[c] - m1[c] * a;
  float d = g2[c] * rsqrtf(v2[c] + 1e-5f);
  s2[c] = d;
  t2[c] = b2[c] - m2[c] * d;
}

__global__ void k_f32_to_bf16(const float* __restrict__ s,
                              __bf16* __restrict__ d, int n) {
  int i = blockIdx.x * 256 + threadIdx.x;
  if (i < n) d[i] = (__bf16)s[i];
}

// red_w is [C][L*C]; repack to [l*C+o][c] so GEMM rows 0..255 = block l=0 etc.
__global__ void k_perm_red(const float* __restrict__ src,
                           __bf16* __restrict__ dst) {
  int i = blockIdx.x * 256 + threadIdx.x;  // (l*256+o)*256 + c
  int c = i & 255, o = (i >> 8) & 255, l = i >> 16;
  dst[i] = (__bf16)src[o * 768 + l * 256 + c];
}

// BN1 + mask logits + 3-way softmax, fused single pass over x.
__global__ __launch_bounds__(256) void k_bn_mask(
    const float* __restrict__ x, const float* __restrict__ convw,
    const float* __restrict__ s1, const float* __restrict__ t1,
    __bf16* __restrict__ xb, float* __restrict__ mask) {
  __shared__ float scw[768], ss[256], st[256];
  int tid = threadIdx.x;
  for (int i = tid; i < 768; i += 256) scw[i] = convw[i];
  ss[tid] = s1[tid];
  st[tid] = t1[tid];
  __syncthreads();

  int n = blockIdx.x * 256 + tid;
  int b = n >> 12, pix = n & 4095;
  size_t base = (size_t)b * CC * PIX + pix;
  float m0 = 0.f, m1 = 0.f, m2 = 0.f;
  for (int c = 0; c < CC; ++c) {
    float v = x[base + (size_t)c * PIX] * ss[c] + st[c];
    xb[base + (size_t)c * PIX] = (__bf16)v;
    m0 += scw[c] * v;
    m1 += scw[256 + c] * v;
    m2 += scw[512 + c] * v;
  }
  float mx = fmaxf(m0, fmaxf(m1, m2));
  float e0 = __expf(m0 - mx), e1 = __expf(m1 - mx), e2 = __expf(m2 - mx);
  float inv = 1.f / (e0 + e1 + e2);
  size_t mb = (size_t)b * 3 * PIX + pix;
  mask[mb] = e0 * inv;
  mask[mb + PIX] = e1 * inv;
  mask[mb + 2 * PIX] = e2 * inv;
}

// Jred[b][o][pix] = sum_l mask[b][l][pix] * Y[b][l*256+o][pix]
__global__ void k_combine(const __bf16* __restrict__ Y,
                          const float* __restrict__ mask,
                          __bf16* __restrict__ J) {
  int i = blockIdx.x * 256 + threadIdx.x;  // b*2^20 + o*4096 + pix
  int pix = i & 4095, o = (i >> 12) & 255, b = i >> 20;
  size_t yb = (size_t)b * 768 * PIX;
  size_t mb = (size_t)b * 3 * PIX + pix;
  float acc = mask[mb] * (float)Y[yb + (size_t)o * PIX + pix];
  acc += mask[mb + PIX] * (float)Y[yb + (size_t)(256 + o) * PIX + pix];
  acc += mask[mb + 2 * PIX] * (float)Y[yb + (size_t)(512 + o) * PIX + pix];
  J[i] = (__bf16)acc;
}

// ---------------------------------------------------------------------------
// Criss-cross attention
// ---------------------------------------------------------------------------
// Scores: per (b, fixed-coord f): E[p][p'] = sum_c q[c,p] * k[c,p'] ; 64x64x32
// eH: f = w (fStride=1, posStride=64); eW: f = h (fStride=64, posStride=1).
__global__ __launch_bounds__(128) void k_attn_scores(
    const __bf16* __restrict__ Q, const __bf16* __restrict__ K,
    float* __restrict__ E, int posStride, int fStride) {
  __shared__ __bf16 Qa[64 * 32];  // [p][c]  (A rows)
  __shared__ __bf16 Kb[64 * 32];  // [p'][c] (B n-major)
  int tid = threadIdx.x, lane = tid & 31, wv = tid >> 5;
  int bid = blockIdx.x, b = bid >> 6, f = bid & 63;
  size_t qb = (size_t)b * C8 * PIX + (size_t)f * fStride;
  for (int idx = tid; idx < 2048; idx += 128) {
    int p = idx >> 5, c = idx & 31;
    size_t g = qb + (size_t)c * PIX + (size_t)p * posStride;
    Qa[idx] = Q[g];
    Kb[idx] = K[g];
  }
  __syncthreads();

  v16bf a = frag_a(Qa + wv * 16 * 32, lane);  // K = 32 = one WMMA step
  int hf = lane >> 4, nl = lane & 15;
#pragma unroll
  for (int j = 0; j < 4; ++j) {
    v16bf bf = frag_b(Kb + j * 16 * 32, lane);
    v8f acc = {0.f, 0.f, 0.f, 0.f, 0.f, 0.f, 0.f, 0.f};
    acc = wmma_bf16(a, bf, acc);
#pragma unroll
    for (int v = 0; v < 8; ++v) {
      int row = wv * 16 + hf * 8 + v, col = j * 16 + nl;
      E[((size_t)bid * 64 + row) * 64 + col] = acc[v];
    }
  }
}

// Joint softmax over the 128 concatenated [eH, eW] logits (diag of eH = -inf).
__global__ void k_attn_softmax(const float* __restrict__ EH,
                               const float* __restrict__ EW,
                               __bf16* __restrict__ AH,
                               __bf16* __restrict__ AW) {
  int n = blockIdx.x * 256 + threadIdx.x;  // 0..NTOT-1
  int b = n >> 12, pix = n & 4095, h = pix >> 6, w = pix & 63;
  const float* eh = EH + (((size_t)(b * 64 + w) * 64 + h) << 6);
  const float* ew = EW + (((size_t)(b * 64 + h) * 64 + w) << 6);
  float mx = -3.4e38f;
  for (int g = 0; g < 64; ++g)
    if (g != h) mx = fmaxf(mx, eh[g]);
  for (int v = 0; v < 64; ++v) mx = fmaxf(mx, ew[v]);
  float s = 0.f;
  for (int g = 0; g < 64; ++g)
    if (g != h) s += __expf(eh[g] - mx);
  for (int v = 0; v < 64; ++v) s += __expf(ew[v] - mx);
  float inv = 1.f / s;
  __bf16* ah = AH + (((size_t)(b * 64 + w) * 64 + h) << 6);  // [b][w][h][g]
  __bf16* aw = AW + (((size_t)(b * 64 + h) * 64 + w) << 6);  // [b][h][w][v]
  for (int g = 0; g < 64; ++g)
    ah[g] = (__bf16)((g == h) ? 0.f : __expf(eh[g] - mx) * inv);
  for (int v = 0; v < 64; ++v) aw[v] = (__bf16)(__expf(ew[v] - mx) * inv);
}

// Attention output GEMM: per (b,f): out[c][p] = sum_k v[c,k] * A[p][k]
// (M=256, N=64, K=64).  PHASE 0 stores outH into Ptmp; PHASE 1 adds outW,
// applies gamma, residual prior, BN2; writes fp32 P and bf16 P for the FFN.
template <int PHASE>
__global__ __launch_bounds__(256) void k_attn_out(
    const __bf16* __restrict__ V, const __bf16* __restrict__ Aat,
    float* __restrict__ Ptmp, float* __restrict__ Pout,
    __bf16* __restrict__ Pbf, const float* __restrict__ prior,
    const float* __restrict__ gamma, const float* __restrict__ s2,
    const float* __restrict__ t2, int posStride, int fStride) {
  __shared__ __bf16 sA[256 * 32];  // [c][k] slab
  __shared__ __bf16 sB[64 * 32];   // [n][k]
  int tid = threadIdx.x, lane = tid & 31, wv = tid >> 5;
  int bid = blockIdx.x, b = bid >> 6, f = bid & 63;
  size_t vbase = (size_t)b * CC * PIX + (size_t)f * fStride;

  v8f acc[2][4];
#pragma unroll
  for (int i = 0; i < 2; ++i)
#pragma unroll
    for (int j = 0; j < 4; ++j) {
      v8f z = {0.f, 0.f, 0.f, 0.f, 0.f, 0.f, 0.f, 0.f};
      acc[i][j] = z;
    }

  for (int kk = 0; kk < 64; kk += 32) {
    for (int idx = tid; idx < 8192; idx += 256) {
      int m = idx >> 5, k = idx & 31;
      sA[idx] = V[vbase + (size_t)m * PIX + (size_t)(kk + k) * posStride];
    }
    {
      int idx = tid * 8;  // 2048 elems, one 16B chunk per thread
      int n = idx >> 5, k = idx & 31;
      *(v8bf*)(sB + idx) =
          *(const v8bf*)(Aat + ((size_t)bid * 64 + n) * 64 + kk + k);
    }
    __syncthreads();
    v16bf aF[2];
#pragma unroll
    for (int i = 0; i < 2; ++i)
      aF[i] = frag_a(sA + (wv * 32 + i * 16) * 32, lane);
#pragma unroll
    for (int j = 0; j < 4; ++j) {
      v16bf bF = frag_b(sB + j * 16 * 32, lane);
#pragma unroll
      for (int i = 0; i < 2; ++i) acc[i][j] = wmma_bf16(aF[i], bF, acc[i][j]);
    }
    __syncthreads();
  }

  float gm = gamma[0];
  int hf = lane >> 4, nl = lane & 15;
#pragma unroll
  for (int i = 0; i < 2; ++i)
#pragma unroll
    for (int j = 0; j < 4; ++j)
#pragma unroll
      for (int v = 0; v < 8; ++v) {
        int m = wv * 32 + i * 16 + hf * 8 + v;
        int n = j * 16 + nl;
        size_t addr = (size_t)(b * CC + m) * PIX + (size_t)n * posStride +
                      (size_t)f * fStride;
        if constexpr (PHASE == 0) {
          Ptmp[addr] = acc[i][j][v];
        } else {
          float s = acc[i][j][v] + Ptmp[addr];
          float pv = gm * s + prior[addr];
          float pbn = pv * s2[m] + t2[m];   // p = BN2(cca out); residual uses BN'd p
          Pout[addr] = pbn;
          Pbf[addr] = (__bf16)pbn;
        }
      }
}

// Depthwise 3x3 SAME conv (+ bias), bf16 in/out, fp32 accumulate.
__global__ void k_dwconv(const __bf16* __restrict__ h1,
                         const float* __restrict__ dww,
                         const float* __restrict__ dwb,
                         __bf16* __restrict__ h2) {
  int i = blockIdx.x * 256 + threadIdx.x;  // b*2^22 + c*4096 + pix
  int pix = i & 4095, c = (i >> 12) & 1023, b = i >> 22;
  int h = pix >> 6, w = pix & 63;
  const __bf16* src = h1 + (size_t)(b * HID + c) * PIX;
  float acc = dwb[c];
#pragma unroll
  for (int di = -1; di <= 1; ++di) {
    int hh = h + di;
    if (hh < 0 || hh > 63) continue;
#pragma unroll
    for (int dj = -1; dj <= 1; ++dj) {
      int ww = w + dj;
      if (ww < 0 || ww > 63) continue;
      acc += dww[c * 9 + (di + 1) * 3 + (dj + 1)] * (float)src[hh * 64 + ww];
    }
  }
  h2[i] = (__bf16)acc;
}

// ---------------------------------------------------------------------------
// Host launcher
// ---------------------------------------------------------------------------
extern "C" void kernel_launch(void* const* d_in, const int* in_sizes, int n_in,
                              void* d_out, int out_size, void* d_ws,
                              size_t ws_size, hipStream_t stream) {
  const float* x       = (const float*)d_in[0];
  const float* prior   = (const float*)d_in[1];
  const float* bn1_g   = (const float*)d_in[2];
  const float* bn1_b   = (const float*)d_in[3];
  const float* bn1_m   = (const float*)d_in[4];
  const float* bn1_v   = (const float*)d_in[5];
  const float* conv_w  = (const float*)d_in[6];
  const float* align_w = (const float*)d_in[7];
  const float* red_w   = (const float*)d_in[8];
  const float* q_w     = (const float*)d_in[9];
  const float* q_b     = (const float*)d_in[10];
  const float* k_w     = (const float*)d_in[11];
  const float* k_b     = (const float*)d_in[12];
  const float* v_w     = (const float*)d_in[13];
  const float* v_b     = (const float*)d_in[14];
  const float* gamma   = (const float*)d_in[15];
  const float* bn2_g   = (const float*)d_in[16];
  const float* bn2_b   = (const float*)d_in[17];
  const float* bn2_m   = (const float*)d_in[18];
  const float* bn2_v   = (const float*)d_in[19];
  const float* ffn_w1  = (const float*)d_in[20];
  const float* ffn_b1  = (const float*)d_in[21];
  const float* dw_w    = (const float*)d_in[22];
  const float* dw_b    = (const float*)d_in[23];
  const float* pw_w    = (const float*)d_in[24];
  const float* pw_b    = (const float*)d_in[25];
  const float* ffn_w2  = (const float*)d_in[26];
  const float* ffn_b2  = (const float*)d_in[27];

  char* base = (char*)d_ws;
  size_t off = 0;
  auto alloc = [&](size_t bytes) -> void* {
    off = (off + 255) & ~(size_t)255;
    void* p = base + off;
    off += bytes;
    return p;
  };

  float*  s1    = (float*)alloc(256 * 4);
  float*  t1    = (float*)alloc(256 * 4);
  float*  s2    = (float*)alloc(256 * 4);
  float*  t2    = (float*)alloc(256 * 4);
  __bf16* wbAl  = (__bf16*)alloc((size_t)256 * 256 * 2);
  __bf16* wbRed = (__bf16*)alloc((size_t)768 * 256 * 2);
  __bf16* wbQ   = (__bf16*)alloc((size_t)32 * 256 * 2);
  __bf16* wbK   = (__bf16*)alloc((size_t)32 * 256 * 2);
  __bf16* wbV   = (__bf16*)alloc((size_t)256 * 256 * 2);
  __bf16* wbW1  = (__bf16*)alloc((size_t)1024 * 256 * 2);
  __bf16* wbPW  = (__bf16*)alloc((size_t)1024 * 1024 * 2);
  __bf16* wbW2  = (__bf16*)alloc((size_t)256 * 1024 * 2);
  __bf16* xbb   = (__bf16*)alloc((size_t)BB * CC * PIX * 2);
  __bf16* prb   = (__bf16*)alloc((size_t)BB * CC * PIX * 2);
  float*  mask  = (float*)alloc((size_t)BB * 3 * PIX * 4);
  __bf16* feat  = (__bf16*)alloc((size_t)BB * CC * PIX * 2);
  __bf16* Ybuf  = (__bf16*)alloc((size_t)BB * 768 * PIX * 2);
  __bf16* Jred  = (__bf16*)alloc((size_t)BB * CC * PIX * 2);
  __bf16* qbf   = (__bf16*)alloc((size_t)BB * C8 * PIX * 2);
  __bf16* kbf   = (__bf16*)alloc((size_t)BB * C8 * PIX * 2);
  __bf16* vbf   = (__bf16*)alloc((size_t)BB * CC * PIX * 2);
  float*  EH    = (float*)alloc((size_t)BB * 64 * 64 * 64 * 4);
  float*  EW    = (float*)alloc((size_t)BB * 64 * 64 * 64 * 4);
  __bf16* AH    = (__bf16*)alloc((size_t)BB * 64 * 64 * 64 * 2);
  __bf16* AW    = (__bf16*)alloc((size_t)BB * 64 * 64 * 64 * 2);
  float*  Ptmp  = (float*)alloc((size_t)BB * CC * PIX * 4);
  float*  Pful  = (float*)alloc((size_t)BB * CC * PIX * 4);
  __bf16* pbb   = (__bf16*)alloc((size_t)BB * CC * PIX * 2);
  __bf16* h1    = (__bf16*)alloc((size_t)BB * HID * PIX * 2);
  __bf16* h2    = (__bf16*)alloc((size_t)BB * HID * PIX * 2);
  __bf16* h3    = (__bf16*)alloc((size_t)BB * HID * PIX * 2);

  auto cvt = [&](const float* s, __bf16* d, int n) {
    k_f32_to_bf16<<<(n + 255) / 256, 256, 0, stream>>>(s, d, n);
  };

  // --- prep ---
  k_prep_bn<<<1, 256, 0, stream>>>(bn1_g, bn1_b, bn1_m, bn1_v, bn2_g, bn2_b,
                                   bn2_m, bn2_v, s1, t1, s2, t2);
  cvt(align_w, wbAl, 256 * 256);
  k_perm_red<<<768, 256, 0, stream>>>(red_w, wbRed);
  cvt(q_w, wbQ, 32 * 256);
  cvt(k_w, wbK, 32 * 256);
  cvt(v_w, wbV, 256 * 256);
  cvt(ffn_w1, wbW1, 1024 * 256);
  cvt(pw_w, wbPW, 1024 * 1024);
  cvt(ffn_w2, wbW2, 256 * 1024);
  cvt(prior, prb, BB * CC * PIX);

  // --- BN1 + mask softmax (fused) ---
  k_bn_mask<<<NTOT / 256, 256, 0, stream>>>(x, conv_w, s1, t1, xbb, mask);

  // --- feat = align_w @ xb ---
  gemm_pix<128, 128, 4, 2, 0><<<dim3(NTOT / 128, 256 / 128), 256, 0, stream>>>(
      wbAl, 256, 256, xbb, nullptr, feat, nullptr, nullptr);
  // --- Y = red3 @ feat  (768 rows = 3 stacked 256x256 blocks) ---
  gemm_pix<128, 128, 4, 2, 0><<<dim3(NTOT / 128, 768 / 128), 256, 0, stream>>>(
      wbRed, 768, 256, feat, nullptr, Ybuf, nullptr, nullptr);
  // --- Jred = sum_l mask_l * Y_l ---
  k_combine<<<(BB * CC * PIX) / 256, 256, 0, stream>>>(Ybuf, mask, Jred);

  // --- q, k, v projections ---
  gemm_pix<32, 256, 1, 8, 0><<<dim3(NTOT / 256, 1), 256, 0, stream>>>(
      wbQ, 32, 256, prb, q_b, qbf, nullptr, nullptr);
  gemm_pix<32, 256, 1, 8, 0><<<dim3(NTOT / 256, 1), 256, 0, stream>>>(
      wbK, 32, 256, Jred, k_b, kbf, nullptr, nullptr);
  gemm_pix<128, 128, 4, 2, 0><<<dim3(NTOT / 128, 256 / 128), 256, 0, stream>>>(
      wbV, 256, 256, Jred, v_b, vbf, nullptr, nullptr);

  // --- criss-cross attention ---
  k_attn_scores<<<BB * 64, 128, 0, stream>>>(qbf, kbf, EH, /*pos*/ 64, /*f*/ 1);
  k_attn_scores<<<BB * 64, 128, 0, stream>>>(qbf, kbf, EW, /*pos*/ 1, /*f*/ 64);
  k_attn_softmax<<<NTOT / 256, 256, 0, stream>>>(EH, EW, AH, AW);
  k_attn_out<0><<<BB * 64, 256, 0, stream>>>(vbf, AH, Ptmp, Pful, pbb, prior,
                                             gamma, s2, t2, 64, 1);
  k_attn_out<1><<<BB * 64, 256, 0, stream>>>(vbf, AW, Ptmp, Pful, pbb, prior,
                                             gamma, s2, t2, 1, 64);

  // --- ConvFFN ---
  gemm_pix<128, 128, 4, 2, 0><<<dim3(NTOT / 128, 1024 / 128), 256, 0, stream>>>(
      wbW1, 1024, 256, pbb, ffn_b1, h1, nullptr, nullptr);
  k_dwconv<<<(BB * HID * PIX) / 256, 256, 0, stream>>>(h1, dw_w, dw_b, h2);
  gemm_pix<128, 128, 4, 2, 1><<<dim3(NTOT / 128, 1024 / 128), 256, 0, stream>>>(
      wbPW, 1024, 1024, h2, pw_b, h3, nullptr, nullptr);
  gemm_pix<128, 128, 4, 2, 2><<<dim3(NTOT / 128, 256 / 128), 256, 0, stream>>>(
      wbW2, 256, 1024, h3, ffn_b2, nullptr, (float*)d_out, Pful);
}
